// InstrumentedTNP_26061861552302
// MI455X (gfx1250) — matrix operand order repaired and checked by reading, not compiled, not run on hardware
//
#include <hip/hip_runtime.h>
#include <hip/hip_bf16.h>
#include <cmath>

// ---------------------------------------------------------------------------
// Model dims (compile-time constants from the reference)
// ---------------------------------------------------------------------------
#define TD    1024
#define NH    16
#define NLAY  6
#define PROT  1280
#define DRUG  2048
#define NCTX  512
#define NQRY  1024
#define NTOK  (NCTX + NQRY)   // 1536
#define HD    (TD / NH)       // 64

typedef __attribute__((ext_vector_type(16))) __bf16 v16bf;
typedef __attribute__((ext_vector_type(8)))  float  v8f;

union FragBF {
    v16bf v;
    unsigned int u[8];
};

__device__ __forceinline__ v8f zero8() {
    v8f z = {0.f, 0.f, 0.f, 0.f, 0.f, 0.f, 0.f, 0.f};
    return z;
}

// fp32 -> bf16 through the native convert (backend picks v_cvt for gfx1250)
__device__ __forceinline__ unsigned short f2bf(float x) {
    __bf16 h = (__bf16)x;
    return __builtin_bit_cast(unsigned short, h);
}

// ---------------------------------------------------------------------------
// WMMA fragment loaders (bf16, per CDNA5 ISA 7.12.2 layouts, wave32)
// A-matrix 16x32: lane L holds row M=L%16; K pattern depends on lane half.
// lds points at (M=0,K=0) of the 16x32 sub-block; ldk = row stride in ushorts.
// ---------------------------------------------------------------------------
__device__ __forceinline__ v16bf frag_a(const unsigned short* lds, int ldk, int lane) {
    FragBF f;
    const int m  = lane & 15;
    const int kb = (lane >> 4) * 8;
    const unsigned short* row = lds + m * ldk;
#pragma unroll
    for (int hf = 0; hf < 2; ++hf) {
        const int kbase = hf * 16 + kb;
#pragma unroll
        for (int p = 0; p < 4; ++p) {
            f.u[hf * 4 + p] = *(const unsigned int*)(row + kbase + 2 * p);
        }
    }
    return f.v;
}

// B-matrix 32x16: lane L holds column N=L%16; K = (L/16)*8 + e (e<8), 16 + (L/16)*8 + (e-8) (e>=8).
// Expect B staged TRANSPOSED in LDS as [N][K] (ldk = K stride in ushorts),
// so each lane reads two runs of 8 contiguous bf16 (k = 8h.., 16+8h..).
__device__ __forceinline__ v16bf frag_b(const unsigned short* lds, int ldk, int lane) {
    FragBF f;
    const int n     = lane & 15;
    const int kbase = (lane >> 4) * 8;
    const unsigned short* p = lds + n * ldk + kbase;
#pragma unroll
    for (int i = 0; i < 8; ++i) f.u[i] = *(const unsigned int*)(p + (i >> 2) * 16 + 2 * (i & 3));
    return f.v;
}

__device__ __forceinline__ v8f wmma_bf16(v16bf a, v16bf b, v8f c) {
    return __builtin_amdgcn_wmma_f32_16x16x32_bf16(
        /*neg_a=*/false, a, /*neg_b=*/false, b,
        /*c_mod=*/(short)0, c, /*reuse_a=*/false, /*reuse_b=*/false);
}

// ---------------------------------------------------------------------------
// Generic tiled GEMM:  C[M,N] = epilogue(A[M,K] @ B[K,N] + bias, res)
// 128x128 tile per 256-thread workgroup (8 waves; each wave 32x64 = 2x4 WMMA
// tiles). fp32 inputs software-pipelined through registers, converted to bf16
// into LDS, f32 accumulation in WMMA. All M,N here are multiples of 128 and
// K multiples of 32 — no bounds checks.
// ---------------------------------------------------------------------------
enum { EPI_BIAS = 0, EPI_RES = 1, EPI_GELU = 2, EPI_RELU = 3 };

template <int EPI>
__global__ __launch_bounds__(256)
void gemm_bf16_wmma(const float* __restrict__ A, const float* __restrict__ B,
                    const float* __restrict__ bias, const float* __restrict__ res,
                    float* __restrict__ C, int M, int N, int K,
                    int lda, int ldb, int ldc) {
    __shared__ unsigned short As[128 * 34];   // [m][k] row-major, padded
    __shared__ unsigned short Bs[128 * 34];   // [n][k] transposed, padded

    const int tid  = threadIdx.x;
    const int lane = tid & 31;
    const int wave = tid >> 5;
    const int m0 = blockIdx.y * 128;
    const int n0 = blockIdx.x * 128;
    const int wm = (wave >> 1) * 32;   // 4 M-slots
    const int wn = (wave & 1) * 64;    // 2 N-slots

    v8f acc[2][4];
#pragma unroll
    for (int i = 0; i < 2; ++i)
#pragma unroll
        for (int j = 0; j < 4; ++j) acc[i][j] = zero8();

    // staging maps (4 float4 per thread for each of A and B)
    const int arow = tid >> 3;          // 0..31 (+32*t)
    const int ac4  = (tid & 7) * 4;     // 0..28
    const int brow = tid >> 5;          // 0..7  (+8*t)
    const int bc4  = (tid & 31) * 4;    // 0..124

    float4 aR[4], bR[4];
    auto load_regs = [&](int k0) {
#pragma unroll
        for (int t = 0; t < 4; ++t)
            aR[t] = *(const float4*)(A + (size_t)(m0 + arow + 32 * t) * lda + k0 + ac4);
#pragma unroll
        for (int t = 0; t < 4; ++t)
            bR[t] = *(const float4*)(B + (size_t)(k0 + brow + 8 * t) * ldb + n0 + bc4);
    };

    load_regs(0);
    for (int k0 = 0; k0 < K; k0 += 32) {
        __syncthreads();   // previous iteration's fragment reads are done
#pragma unroll
        for (int t = 0; t < 4; ++t) {
            unsigned short* dst = As + (arow + 32 * t) * 34 + ac4;
            dst[0] = f2bf(aR[t].x); dst[1] = f2bf(aR[t].y);
            dst[2] = f2bf(aR[t].z); dst[3] = f2bf(aR[t].w);
        }
#pragma unroll
        for (int t = 0; t < 4; ++t) {
            const int k = brow + 8 * t;
            Bs[(bc4 + 0) * 34 + k] = f2bf(bR[t].x);
            Bs[(bc4 + 1) * 34 + k] = f2bf(bR[t].y);
            Bs[(bc4 + 2) * 34 + k] = f2bf(bR[t].z);
            Bs[(bc4 + 3) * 34 + k] = f2bf(bR[t].w);
        }
        __syncthreads();

        // overlap next slab's global loads with this slab's WMMAs
        if (k0 + 32 < K) {
            __builtin_prefetch(A + (size_t)(m0 + arow) * lda + k0 + 64, 0, 1);
            load_regs(k0 + 32);
        }

        v16bf afr[2];
#pragma unroll
        for (int i = 0; i < 2; ++i)
            afr[i] = frag_a(As + (wm + i * 16) * 34, 34, lane);
#pragma unroll
        for (int j = 0; j < 4; ++j) {
            const v16bf b = frag_b(Bs + (wn + j * 16) * 34, 34, lane);
            acc[0][j] = wmma_bf16(afr[0], b, acc[0][j]);
            acc[1][j] = wmma_bf16(afr[1], b, acc[1][j]);
        }
    }

    // epilogue: D layout -> lane holds (M = r + 8*(lane/16), N = lane%16) per VGPR r
    const int nc   = lane & 15;
    const int hsel = lane >> 4;
#pragma unroll
    for (int i = 0; i < 2; ++i)
#pragma unroll
        for (int j = 0; j < 4; ++j)
#pragma unroll
            for (int r = 0; r < 8; ++r) {
                const int row = m0 + wm + i * 16 + r + 8 * hsel;
                const int col = n0 + wn + j * 16 + nc;
                float c = acc[i][j][r] + bias[col];
                if constexpr (EPI == EPI_RES)  c += res[(size_t)row * ldc + col];
                if constexpr (EPI == EPI_GELU) c = 0.5f * c * (1.f + erff(c * 0.70710678118654752f));
                if constexpr (EPI == EPI_RELU) c = fmaxf(c, 0.f);
                C[(size_t)row * ldc + col] = c;
            }
}

// ---------------------------------------------------------------------------
// Flash attention, one wave per (query-tile of 16, head).
// qkv: [NTOK][3*TD] (q | k | v). out: [NTOK][TD] concatenated heads.
// ---------------------------------------------------------------------------
__device__ __forceinline__ bool attn_masked(int i, int j) {
    return ((i >= NCTX) && (j >= NCTX) && (i != j)) || ((i < NCTX) && (j >= NCTX));
}

__global__ __launch_bounds__(32)
void attn_kernel(const float* __restrict__ qkv,
                 const float* __restrict__ ppr, const float* __restrict__ trust,
                 const float* __restrict__ alpha_p, const float* __restrict__ tscale_p,
                 int layer, float* __restrict__ out) {
    __shared__ unsigned short Qs[16 * 64];
    __shared__ unsigned short Ks[32 * 64];   // [key][hd]
    __shared__ unsigned short Vt[64 * 32];   // [hd][key] (transposed for B frags)
    __shared__ unsigned short Ps[16 * 32];   // softmax probs as A fragment source
    __shared__ float biasS[32];

    const int lane = threadIdx.x;
    const int q0   = blockIdx.x * 16;
    const int h    = blockIdx.y;
    const float alpha  = alpha_p[layer];
    const float tscale = tscale_p[layer];
    const float qscale = 0.125f;            // 1/sqrt(HD)

    // stage Q tile (pre-scaled)
    for (int i = lane; i < 16 * 64; i += 32) {
        const int r = i >> 6, c = i & 63;
        Qs[i] = f2bf(qkv[(size_t)(q0 + r) * (3 * TD) + h * HD + c] * qscale);
    }

    v8f o0 = zero8(), o1 = zero8(), o2 = zero8(), o3 = zero8();
    float rm[8], rl[8];
#pragma unroll
    for (int r = 0; r < 8; ++r) { rm[r] = -1e30f; rl[r] = 0.f; }

    const int ncol = lane & 15;
    const int hsel = lane >> 4;

    for (int kc = 0; kc < NTOK / 32; ++kc) {
        __syncthreads();
        const int k0 = kc * 32;
        // stage K chunk (natural) and trust-gated V chunk (transposed)
        for (int i = lane; i < 32 * 64; i += 32) {
            const int r = i >> 6, c = i & 63;
            const int j = k0 + r;
            Ks[i] = f2bf(qkv[(size_t)j * (3 * TD) + TD + h * HD + c]);
            const float g = (j < NCTX) ? 1.f / (1.f + expf(-tscale * trust[j])) : 1.f;
            Vt[c * 32 + r] = f2bf(qkv[(size_t)j * (3 * TD) + 2 * TD + h * HD + c] * g);
        }
        {
            const int j = k0 + lane;
            biasS[lane] = (j < NCTX) ? alpha * logf(fmaxf(ppr[j], 1e-8f)) : 0.f;
        }
        __syncthreads();

        // S = Q @ K^T : two 16x16 tiles, K-dim = 64 (2 WMMA steps)
        v8f s0 = zero8(), s1 = zero8();
#pragma unroll
        for (int kk = 0; kk < 2; ++kk) {
            const v16bf a  = frag_a(Qs + kk * 32, 64, lane);
            const v16bf b0 = frag_b(Ks + 0 * 16 * 64 + kk * 32, 64, lane);
            const v16bf b1 = frag_b(Ks + 1 * 16 * 64 + kk * 32, 64, lane);
            s0 = wmma_bf16(a, b0, s0);
            s1 = wmma_bf16(a, b1, s1);
        }

        // bias + mask + online softmax update
#pragma unroll
        for (int r = 0; r < 8; ++r) {
            const int m  = r + 8 * hsel;
            const int qi = q0 + m;
            const int j0 = k0 + ncol;
            const int j1 = k0 + 16 + ncol;
            float sv0 = attn_masked(qi, j0) ? -1e30f : (s0[r] + biasS[ncol]);
            float sv1 = attn_masked(qi, j1) ? -1e30f : (s1[r] + biasS[16 + ncol]);

            float mx = fmaxf(sv0, sv1);
#pragma unroll
            for (int off = 1; off < 16; off <<= 1)
                mx = fmaxf(mx, __shfl_xor(mx, off, 32));
            const float newm  = fmaxf(rm[r], mx);
            const float scale = expf(rm[r] - newm);
            const float p0 = expf(sv0 - newm);
            const float p1 = expf(sv1 - newm);
            float ps = p0 + p1;
#pragma unroll
            for (int off = 1; off < 16; off <<= 1)
                ps += __shfl_xor(ps, off, 32);
            rl[r] = rl[r] * scale + ps;
            rm[r] = newm;
            o0[r] *= scale; o1[r] *= scale; o2[r] *= scale; o3[r] *= scale;
            Ps[m * 32 + ncol]      = f2bf(p0);
            Ps[m * 32 + 16 + ncol] = f2bf(p1);
        }
        __syncthreads();

        // O += P @ V  (4 column tiles of HD=64)
        const v16bf pa = frag_a(Ps, 32, lane);
        o0 = wmma_bf16(pa, frag_b(Vt + 0 * 16 * 32, 32, lane), o0);
        o1 = wmma_bf16(pa, frag_b(Vt + 1 * 16 * 32, 32, lane), o1);
        o2 = wmma_bf16(pa, frag_b(Vt + 2 * 16 * 32, 32, lane), o2);
        o3 = wmma_bf16(pa, frag_b(Vt + 3 * 16 * 32, 32, lane), o3);
    }

    // normalize and write
#pragma unroll
    for (int r = 0; r < 8; ++r) {
        const int m = r + 8 * hsel;
        const float inv = 1.f / rl[r];
        float* orow = out + (size_t)(q0 + m) * TD + h * HD;
        orow[0 * 16 + ncol] = o0[r] * inv;
        orow[1 * 16 + ncol] = o1[r] * inv;
        orow[2 * 16 + ncol] = o2[r] * inv;
        orow[3 * 16 + ncol] = o3[r] * inv;
    }
}

// ---------------------------------------------------------------------------
// LayerNorm over rows of [*, TD]
// ---------------------------------------------------------------------------
__global__ __launch_bounds__(256)
void layernorm_kernel(const float* __restrict__ x, const float* __restrict__ g,
                      const float* __restrict__ b, float* __restrict__ y) {
    __shared__ float red[256];
    const int row = blockIdx.x;
    const int tid = threadIdx.x;
    const float* xr = x + (size_t)row * TD;

    float s = 0.f;
    for (int c = tid; c < TD; c += 256) s += xr[c];
    red[tid] = s; __syncthreads();
    for (int o = 128; o > 0; o >>= 1) { if (tid < o) red[tid] += red[tid + o]; __syncthreads(); }
    const float mean = red[0] * (1.f / TD);
    __syncthreads();

    float v = 0.f;
    for (int c = tid; c < TD; c += 256) { const float d = xr[c] - mean; v += d * d; }
    red[tid] = v; __syncthreads();
    for (int o = 128; o > 0; o >>= 1) { if (tid < o) red[tid] += red[tid + o]; __syncthreads(); }
    const float inv = rsqrtf(red[0] * (1.f / TD) + 1e-5f);

    float* yr = y + (size_t)row * TD;
    for (int c = tid; c < TD; c += 256)
        yr[c] = (xr[c] - mean) * inv * g[c] + b[c];
}

// X[M,N] += v1[col] + scale*v2[col] + v3[col]   (v2/v3 optional)
__global__ void addrow_kernel(float* __restrict__ X, int total, int N,
                              const float* __restrict__ v1,
                              const float* __restrict__ v2, float scale,
                              const float* __restrict__ v3) {
    const int idx = blockIdx.x * blockDim.x + threadIdx.x;
    if (idx >= total) return;
    const int col = idx % N;
    float a = v1[col];
    if (v2) a += scale * v2[col];
    if (v3) a += v3[col];
    X[idx] += a;
}

// ctx_a outer product into concat buffer cols [TD, TD+TD/4)
__global__ void ctxa_kernel(const float* __restrict__ a, const float* __restrict__ W,
                            const float* __restrict__ b, float* __restrict__ cat, int ldc) {
    const int idx = blockIdx.x * blockDim.x + threadIdx.x;
    if (idx >= NCTX * (TD / 4)) return;
    const int i = idx / (TD / 4);
    const int j = idx % (TD / 4);
    cat[(size_t)i * ldc + TD + j] = a[i] * W[j] + b[j];
}

// final head: hh[1024,512] @ W2[512,2] + b2 -> mu / softplus(sigma)
__global__ void head_kernel(const float* __restrict__ hh, const float* __restrict__ W2,
                            const float* __restrict__ b2, const float* __restrict__ csb,
                            float density, float* __restrict__ out) {
    const int row = blockIdx.x * blockDim.x + threadIdx.x;
    if (row >= NQRY) return;
    const float* hr = hh + (size_t)row * (TD / 2);
    float m = 0.f, s = 0.f;
    for (int k = 0; k < TD / 2; ++k) {
        const float hv = hr[k];
        m += hv * W2[k * 2 + 0];
        s += hv * W2[k * 2 + 1];
    }
    m += b2[0];
    float ls = s + b2[1] + csb[0] * (1.f - density);
    const float sp = (ls > 20.f) ? ls : log1pf(expf(ls));
    out[row]        = m;
    out[NQRY + row] = sp + 1e-4f;
}

// ---------------------------------------------------------------------------
// Host orchestration
// ---------------------------------------------------------------------------
enum {
    IP_CTX_P_W = 0, IP_CTX_P_B, IP_CTX_D_W, IP_CTX_D_B, IP_CTX_A_W, IP_CTX_A_B,
    IP_CTX_F_W, IP_CTX_F_B, IP_QRY_P_W, IP_QRY_P_B, IP_QRY_D_W, IP_QRY_D_B,
    IP_QRY_F_W, IP_QRY_F_B, IP_TYPE_EMBED, IP_DENS_W, IP_DENS_B,
    IP_LN1_G, IP_LN1_B, IP_LN2_G, IP_LN2_B, IP_WQKV, IP_BQKV, IP_WO, IP_BO,
    IP_ALPHA, IP_TSCALE, IP_W1, IP_B1, IP_W2, IP_B2, IP_FN_G, IP_FN_B,
    IP_HEAD_W1, IP_HEAD_B1, IP_HEAD_W2, IP_HEAD_B2, IP_COLD
};

extern "C" void kernel_launch(void* const* d_in, const int* in_sizes, int n_in,
                              void* d_out, int out_size, void* d_ws, size_t ws_size,
                              hipStream_t stream) {
    const float* ctx_p   = (const float*)d_in[0];
    const float* ctx_d   = (const float*)d_in[1];
    const float* ctx_a   = (const float*)d_in[2];
    const float* qry_p   = (const float*)d_in[3];
    const float* qry_d   = (const float*)d_in[4];
    const float* ctx_ppr = (const float*)d_in[5];
    const float* ctx_tr  = (const float*)d_in[6];
#define PRM(i) ((const float*)d_in[7 + (i)])

    float* ws = (float*)d_ws;
    float* xb     = ws;                      // [1536][1024]
    float* hb     = xb     + (size_t)NTOK * TD;
    float* qkvb   = hb     + (size_t)NTOK * TD;          // [1536][3072]
    float* attnb  = qkvb   + (size_t)NTOK * 3 * TD;      // [1536][1024]
    float* ffh    = attnb  + (size_t)NTOK * TD;          // [1536][2048]
    float* catc   = ffh    + (size_t)NTOK * 2 * TD;      // [512][1280]
    float* catq   = catc   + (size_t)NCTX * (PROT);      // [1024][1024]
    float* headh  = catq   + (size_t)NQRY * TD;          // [1024][512]

    const float density = tanhf((float)NCTX / 32.f);
    const dim3 blk256(256), blk32(32);

    // ---- embeddings ----
    // cp -> catc[:, 0:512]
    gemm_bf16_wmma<EPI_BIAS><<<dim3(512 / 128, NCTX / 128), blk256, 0, stream>>>(
        ctx_p, PRM(IP_CTX_P_W), PRM(IP_CTX_P_B), nullptr, catc,
        NCTX, 512, PROT, PROT, 512, PROT);
    // cd -> catc[:, 512:1024]
    gemm_bf16_wmma<EPI_BIAS><<<dim3(512 / 128, NCTX / 128), blk256, 0, stream>>>(
        ctx_d, PRM(IP_CTX_D_W), PRM(IP_CTX_D_B), nullptr, catc + 512,
        NCTX, 512, DRUG, DRUG, 512, PROT);
    // ca -> catc[:, 1024:1280]
    ctxa_kernel<<<(NCTX * (TD / 4) + 255) / 256, blk256, 0, stream>>>(
        ctx_a, PRM(IP_CTX_A_W), PRM(IP_CTX_A_B), catc, PROT);
    // ct -> xb[0:512]
    gemm_bf16_wmma<EPI_BIAS><<<dim3(TD / 128, NCTX / 128), blk256, 0, stream>>>(
        catc, PRM(IP_CTX_F_W), PRM(IP_CTX_F_B), nullptr, xb,
        NCTX, TD, PROT, PROT, TD, TD);
    addrow_kernel<<<(NCTX * TD + 255) / 256, blk256, 0, stream>>>(
        xb, NCTX * TD, TD, PRM(IP_TYPE_EMBED), nullptr, 0.f, nullptr);

    // qp -> catq[:, 0:512], qd -> catq[:, 512:1024]
    gemm_bf16_wmma<EPI_BIAS><<<dim3(512 / 128, NQRY / 128), blk256, 0, stream>>>(
        qry_p, PRM(IP_QRY_P_W), PRM(IP_QRY_P_B), nullptr, catq,
        NQRY, 512, PROT, PROT, 512, TD);
    gemm_bf16_wmma<EPI_BIAS><<<dim3(512 / 128, NQRY / 128), blk256, 0, stream>>>(
        qry_d, PRM(IP_QRY_D_W), PRM(IP_QRY_D_B), nullptr, catq + 512,
        NQRY, 512, DRUG, DRUG, 512, TD);
    // qt -> xb[512:1536]
    gemm_bf16_wmma<EPI_BIAS><<<dim3(TD / 128, NQRY / 128), blk256, 0, stream>>>(
        catq, PRM(IP_QRY_F_W), PRM(IP_QRY_F_B), nullptr, xb + (size_t)NCTX * TD,
        NQRY, TD, TD, TD, TD, TD);
    addrow_kernel<<<(NQRY * TD + 255) / 256, blk256, 0, stream>>>(
        xb + (size_t)NCTX * TD, NQRY * TD, TD,
        PRM(IP_TYPE_EMBED) + TD, PRM(IP_DENS_W), density, PRM(IP_DENS_B));

    // ---- transformer layers ----
    for (int l = 0; l < NLAY; ++l) {
        layernorm_kernel<<<NTOK, blk256, 0, stream>>>(
            xb, PRM(IP_LN1_G) + (size_t)l * TD, PRM(IP_LN1_B) + (size_t)l * TD, hb);

        gemm_bf16_wmma<EPI_BIAS><<<dim3(3 * TD / 128, NTOK / 128), blk256, 0, stream>>>(
            hb, PRM(IP_WQKV) + (size_t)l * TD * 3 * TD, PRM(IP_BQKV) + (size_t)l * 3 * TD,
            nullptr, qkvb, NTOK, 3 * TD, TD, TD, 3 * TD, 3 * TD);

        attn_kernel<<<dim3(NTOK / 16, NH), blk32, 0, stream>>>(
            qkvb, ctx_ppr, ctx_tr, PRM(IP_ALPHA), PRM(IP_TSCALE), l, attnb);

        gemm_bf16_wmma<EPI_RES><<<dim3(TD / 128, NTOK / 128), blk256, 0, stream>>>(
            attnb, PRM(IP_WO) + (size_t)l * TD * TD, PRM(IP_BO) + (size_t)l * TD,
            xb, xb, NTOK, TD, TD, TD, TD, TD);

        layernorm_kernel<<<NTOK, blk256, 0, stream>>>(
            xb, PRM(IP_LN2_G) + (size_t)l * TD, PRM(IP_LN2_B) + (size_t)l * TD, hb);

        gemm_bf16_wmma<EPI_GELU><<<dim3(2 * TD / 128, NTOK / 128), blk256, 0, stream>>>(
            hb, PRM(IP_W1) + (size_t)l * TD * 2 * TD, PRM(IP_B1) + (size_t)l * 2 * TD,
            nullptr, ffh, NTOK, 2 * TD, TD, TD, 2 * TD, 2 * TD);

        gemm_bf16_wmma<EPI_RES><<<dim3(TD / 128, NTOK / 128), blk256, 0, stream>>>(
            ffh, PRM(IP_W2) + (size_t)l * 2 * TD * TD, PRM(IP_B2) + (size_t)l * TD,
            xb, xb, NTOK, TD, 2 * TD, 2 * TD, TD, TD);
    }

    // ---- final norm + head ----
    layernorm_kernel<<<NTOK, blk256, 0, stream>>>(xb, PRM(IP_FN_G), PRM(IP_FN_B), hb);

    gemm_bf16_wmma<EPI_RELU><<<dim3((TD / 2) / 128, NQRY / 128), blk256, 0, stream>>>(
        hb + (size_t)NCTX * TD, PRM(IP_HEAD_W1), PRM(IP_HEAD_B1), nullptr, headh,
        NQRY, TD / 2, TD, TD, TD / 2, TD / 2);

    head_kernel<<<(NQRY + 255) / 256, blk256, 0, stream>>>(
        headh, PRM(IP_HEAD_W2), PRM(IP_HEAD_B2), PRM(IP_COLD), density, (float*)d_out);
#undef PRM
}
